// GNN_4157528343199
// MI455X (gfx1250) — matrix operand order repaired and checked by reading, not compiled, not run on hardware
//
#include <hip/hip_runtime.h>

typedef __attribute__((ext_vector_type(16))) _Float16 v16h;
typedef __attribute__((ext_vector_type(8)))  _Float16 v8h;
typedef __attribute__((ext_vector_type(8)))  float    v8f;
typedef __attribute__((ext_vector_type(4)))  float    v4f;

#define NN 100000          // nodes
#define NE 250000          // edges
#define D  300             // embedding dim
#define BN_EPS 1e-5f

#define AGG_LD  320        // agg [NN,320] f32, K-padded (zeros) for GEMM1
#define HMID_LD 640        // hmid [NN,640] f16, cols 600..639 zero (K-pad for GEMM2)
#define HOUT_LD 320        // hout [NN,320] f32 (20 N-tiles)
#define MT   (NN/16)       // 6250 M-tiles (exact)
#define KC1  (AGG_LD/32)   // 10 K-chunks, GEMM1
#define NT1  (HMID_LD/16)  // 40 N-tiles, GEMM1
#define KC2  (HMID_LD/32)  // 20 K-chunks, GEMM2
#define NT2  (HOUT_LD/16)  // 20 N-tiles, GEMM2 (tiles >= 19 are pure pad)
#define WPB  8             // m-tile waves per GEMM block (share B panel via LDS)
#define GEMM_BLK (WPB*32)  // 256 threads
#define NTG1 8             // N-tiles per wave, GEMM1 (NT1/NTG1 = 5 groups)
#define NTG2 5             // N-tiles per wave, GEMM2 (NT2/NTG2 = 4 groups)
#define PANEL1 (NTG1*KC1*32)   // 2560 v16h = 80 KB LDS
#define PANEL2 (NTG2*KC2*32)   // 3200 v16h = 100 KB LDS

// ---------------------------------------------------------------- utilities
__global__ void zero_f32(float* __restrict__ p, long n) {
    long i = (long)blockIdx.x * blockDim.x + threadIdx.x;
    long stride = (long)gridDim.x * blockDim.x;
    for (; i < n; i += stride) p[i] = 0.f;
}

// ------------------------------------------------------------- atom encoder
__global__ void atom_encoder(const int* __restrict__ x,
                             const float* __restrict__ e1,
                             const float* __restrict__ e2,
                             float* __restrict__ h) {
    long total = (long)NN * D;
    long i = (long)blockIdx.x * blockDim.x + threadIdx.x;
    long stride = (long)gridDim.x * blockDim.x;
    for (; i < total; i += stride) {
        long row = i / D;
        int  col = (int)(i - row * D);
        h[i] = e1[(long)x[2*row] * D + col] + e2[(long)x[2*row+1] * D + col];
    }
}

// -------------------------------------- edge-embedding combination table
// comb[a0*3+a1][d] = edge_emb1[l][a0][d] + edge_emb2[l][a1][d]; 18x300 floats
__global__ void edge_comb(const float* __restrict__ ee1,
                          const float* __restrict__ ee2,
                          float* __restrict__ comb) {
    int i = blockIdx.x * blockDim.x + threadIdx.x;
    if (i >= 18 * D) return;
    int c = i / D, d = i - c * D;
    int a0 = c / 3, a1 = c - a0 * 3;
    comb[i] = ee1[(long)a0 * D + d] + ee2[(long)a1 * D + d];
}

// --------------------------------------------------- message pass (scatter)
// one block per edge (incl. NN self-loops with attr (4,0));
// agg[dst] += h[src] + comb[a0*3+a1]  (f32 non-returning L2 atomics)
__global__ void message_scatter(const int* __restrict__ ei,
                                const int* __restrict__ ea,
                                const float* __restrict__ h,
                                const float* __restrict__ comb,
                                float* __restrict__ agg) {
    int e = blockIdx.x;
    int s, t, c;
    if (e < NE) { s = ei[e]; t = ei[NE + e]; c = ea[2*e] * 3 + ea[2*e+1]; }
    else        { int n = e - NE; s = n; t = n; c = 4 * 3 + 0; }
    const float* hs = h    + (long)s * D;
    const float* pc = comb + (long)c * D;
    float*       ag = agg  + (long)t * AGG_LD;
    int d = threadIdx.x;                       // blockDim = 320, one pass
    if (d < D) unsafeAtomicAdd(&ag[d], hs[d] + pc[d]);
}

// ----------------------------------------- pack W (f32 KxN) into B fragments
// fragment t = ((nt*Kc + kc)*32 + lane): lane holds col n = nt*16+(lane&15),
// K offsets (lane>>4)*16 + j (j=0..15), zero-padded outside [K)x[N).
__global__ void pack_w(const float* __restrict__ W, v16h* __restrict__ out,
                       int K, int N, int Kc, int Nt) {
    int t = blockIdx.x * blockDim.x + threadIdx.x;
    int total = Nt * Kc * 32;
    if (t >= total) return;
    int lane = t & 31;
    int kc   = (t >> 5) % Kc;
    int nt   = t / (Kc * 32);
    int hi = lane >> 4;
    int n  = nt * 16 + (lane & 15);
    v16h v;
#pragma unroll
    for (int j = 0; j < 16; ++j) {
        int k = kc * 32 + hi * 16 + j;
        v[j] = (k < K && n < N) ? (_Float16)W[(long)k * N + n] : (_Float16)0.f;
    }
    out[t] = v;
}

// -------------------------------------------------- GEMM1: agg @ W1 -> hmid
// Block: 8 m-tile waves share one LDS-staged B panel (8 N-tiles x 10 K-chunks).
__global__ __launch_bounds__(GEMM_BLK)
void gemm1_kernel(const float* __restrict__ agg, const v16h* __restrict__ Bp,
                  const float* __restrict__ bias, _Float16* __restrict__ hmid) {
    extern __shared__ v16h sB[];             // PANEL1 entries (80 KB)
    int tid = threadIdx.x;
    int ng  = blockIdx.y;                    // N-tile group [0, NT1/NTG1)
    const v16h* gB = Bp + (long)(ng * NTG1) * KC1 * 32;
    for (int t = tid; t < PANEL1; t += GEMM_BLK) sB[t] = gB[t];
    __syncthreads();

    int lane  = tid & 31;
    int wid   = tid >> 5;
    int mtile = blockIdx.x * WPB + wid;
    if (mtile >= MT) return;                 // wave-uniform exit after barrier
    int hi = lane >> 4, lo = lane & 15;
    const float* arow = agg + (long)(mtile * 16 + lo) * AGG_LD;
    v8f c[NTG1] = {};
    for (int kc = 0; kc < KC1; ++kc) {
        int base = kc * 32 + hi * 16;
        v4f a0 = *(const v4f*)(arow + base);
        v4f a1 = *(const v4f*)(arow + base + 4);
        v4f a2 = *(const v4f*)(arow + base + 8);
        v4f a3 = *(const v4f*)(arow + base + 12);
        v16h a;
#pragma unroll
        for (int j = 0; j < 4; ++j) {
            a[j]      = (_Float16)a0[j];
            a[4 + j]  = (_Float16)a1[j];
            a[8 + j]  = (_Float16)a2[j];
            a[12 + j] = (_Float16)a3[j];
        }
#pragma unroll
        for (int i = 0; i < NTG1; ++i) {
            v16h b = sB[(i * KC1 + kc) * 32 + lane];
            c[i] = __builtin_amdgcn_wmma_f32_16x16x32_f16(false, a, false, b,
                                                          (short)0, c[i], false, false);
        }
    }
#pragma unroll
    for (int i = 0; i < NTG1; ++i) {
        int col = (ng * NTG1 + i) * 16 + lo;
        float bv = (col < 2 * D) ? bias[col] : 0.f;
#pragma unroll
        for (int r = 0; r < 8; ++r) {
            float v = c[i][r] + bv;
            v = v > 0.f ? v : 0.f;
            if (col >= 2 * D) v = 0.f;       // keep K-pad of hmid exactly zero
            hmid[(long)(mtile * 16 + r + 8 * hi) * HMID_LD + col] = (_Float16)v;
        }
    }
}

// ------------------------------------------------- GEMM2: hmid @ W2 -> hout
__global__ __launch_bounds__(GEMM_BLK)
void gemm2_kernel(const _Float16* __restrict__ hmid, const v16h* __restrict__ Bp,
                  const float* __restrict__ bias, float* __restrict__ hout) {
    extern __shared__ v16h sB[];             // PANEL2 entries (100 KB)
    int tid = threadIdx.x;
    int ng  = blockIdx.y;                    // [0, NT2/NTG2)
    const v16h* gB = Bp + (long)(ng * NTG2) * KC2 * 32;
    for (int t = tid; t < PANEL2; t += GEMM_BLK) sB[t] = gB[t];
    __syncthreads();

    int lane  = tid & 31;
    int wid   = tid >> 5;
    int mtile = blockIdx.x * WPB + wid;
    if (mtile >= MT) return;
    int hi = lane >> 4, lo = lane & 15;
    const _Float16* arow = hmid + (long)(mtile * 16 + lo) * HMID_LD;
    v8f c[NTG2] = {};
    for (int kc = 0; kc < KC2; ++kc) {
        int base = kc * 32 + hi * 16;
        v8h a0 = *(const v8h*)(arow + base);
        v8h a1 = *(const v8h*)(arow + base + 8);
        v16h a;
#pragma unroll
        for (int j = 0; j < 8; ++j) { a[j] = a0[j]; a[8 + j] = a1[j]; }
#pragma unroll
        for (int i = 0; i < NTG2; ++i) {
            v16h b = sB[(i * KC2 + kc) * 32 + lane];
            c[i] = __builtin_amdgcn_wmma_f32_16x16x32_f16(false, a, false, b,
                                                          (short)0, c[i], false, false);
        }
    }
#pragma unroll
    for (int i = 0; i < NTG2; ++i) {
        int col = (ng * NTG2 + i) * 16 + lo; // col < 320 always
        float bv = (col < D) ? bias[col] : 0.f;
#pragma unroll
        for (int r = 0; r < 8; ++r)
            hout[(long)(mtile * 16 + r + 8 * hi) * HOUT_LD + col] = c[i][r] + bv;
    }
}

// ------------------------------------------------------------ batch norm
#define BN_ROWS 500   // 200 blocks * 500 rows = 100000
__global__ void bn_stats(const float* __restrict__ hout,
                         float* __restrict__ sums, float* __restrict__ sqs) {
    long r0 = (long)blockIdx.x * BN_ROWS;
    for (int col = threadIdx.x; col < D; col += blockDim.x) {
        float s = 0.f, q = 0.f;
        const float* p = hout + r0 * HOUT_LD + col;
        for (int r = 0; r < BN_ROWS; ++r) {
            float v = p[(long)r * HOUT_LD];
            s += v; q += v * v;
        }
        unsafeAtomicAdd(&sums[col], s);
        unsafeAtomicAdd(&sqs[col], q);
    }
}

__global__ void bn_apply(const float* __restrict__ hout,
                         const float* __restrict__ sums, const float* __restrict__ sqs,
                         const float* __restrict__ gamma, const float* __restrict__ beta,
                         float* __restrict__ h, int do_relu) {
    long total = (long)NN * D;
    long i = (long)blockIdx.x * blockDim.x + threadIdx.x;
    long stride = (long)gridDim.x * blockDim.x;
    const float invN = 1.f / (float)NN;
    for (; i < total; i += stride) {
        long row = i / D;
        int  col = (int)(i - row * D);
        float mean = sums[col] * invN;
        float var  = sqs[col] * invN - mean * mean;
        float v = (hout[row * HOUT_LD + col] - mean) * rsqrtf(var + BN_EPS)
                  * gamma[col] + beta[col];
        if (do_relu) v = fmaxf(v, 0.f);
        h[i] = v;
    }
}

// ---------------------------------------------------------------- launcher
extern "C" void kernel_launch(void* const* d_in, const int* in_sizes, int n_in,
                              void* d_out, int out_size, void* d_ws, size_t ws_size,
                              hipStream_t stream) {
    const int*   x     = (const int*)d_in[0];
    const int*   ei    = (const int*)d_in[1];
    const int*   ea    = (const int*)d_in[2];
    const float* aemb1 = (const float*)d_in[3];
    const float* aemb2 = (const float*)d_in[4];
    const float* eemb1 = (const float*)d_in[5];
    const float* eemb2 = (const float*)d_in[6];
    const float* W1    = (const float*)d_in[7];
    const float* b1    = (const float*)d_in[8];
    const float* W2    = (const float*)d_in[9];
    const float* b2    = (const float*)d_in[10];
    const float* gamma = (const float*)d_in[11];
    const float* beta  = (const float*)d_in[12];
    float* h = (float*)d_out;

    char* ws = (char*)d_ws;
    size_t off = 0;
    auto take = [&](size_t bytes) -> char* {
        char* p = ws + off;
        off += (bytes + 255) & ~(size_t)255;
        return p;
    };
    float*    agg  = (float*)   take((size_t)NN * AGG_LD  * sizeof(float));
    _Float16* hmid = (_Float16*)take((size_t)NN * HMID_LD * sizeof(_Float16));
    float*    hout = (float*)   take((size_t)NN * HOUT_LD * sizeof(float));
    v16h*     pB1  = (v16h*)    take((size_t)NT1 * KC1 * 32 * sizeof(v16h));
    v16h*     pB2  = (v16h*)    take((size_t)NT2 * KC2 * 32 * sizeof(v16h));
    float*    comb = (float*)   take((size_t)18 * D * sizeof(float));
    float*    sums = (float*)   take(HOUT_LD * sizeof(float));
    float*    sqs  = (float*)   take(HOUT_LD * sizeof(float));

    long elems = (long)NN * D;
    int  egrid = (int)((elems + 255) / 256);

    atom_encoder<<<egrid, 256, 0, stream>>>(x, aemb1, aemb2, h);

    dim3 g1((MT + WPB - 1) / WPB, NT1 / NTG1);   // (782, 5)
    dim3 g2((MT + WPB - 1) / WPB, NT2 / NTG2);   // (782, 4)

    for (int l = 0; l < 5; ++l) {
        pack_w<<<(NT1 * KC1 * 32 + 255) / 256, 256, 0, stream>>>(
            W1 + (size_t)l * D * (2 * D), pB1, D, 2 * D, KC1, NT1);
        pack_w<<<(NT2 * KC2 * 32 + 255) / 256, 256, 0, stream>>>(
            W2 + (size_t)l * (2 * D) * D, pB2, 2 * D, D, KC2, NT2);
        edge_comb<<<(18 * D + 255) / 256, 256, 0, stream>>>(
            eemb1 + (size_t)l * 6 * D, eemb2 + (size_t)l * 3 * D, comb);
        zero_f32<<<4096, 256, 0, stream>>>(agg, (long)NN * AGG_LD);
        zero_f32<<<2, 256, 0, stream>>>(sums, HOUT_LD);
        zero_f32<<<2, 256, 0, stream>>>(sqs, HOUT_LD);

        message_scatter<<<NE + NN, 320, 0, stream>>>(ei, ea, h, comb, agg);

        gemm1_kernel<<<g1, GEMM_BLK, PANEL1 * sizeof(v16h), stream>>>(
            agg, pB1, b1 + (size_t)l * 2 * D, hmid);
        gemm2_kernel<<<g2, GEMM_BLK, PANEL2 * sizeof(v16h), stream>>>(
            hmid, pB2, b2 + (size_t)l * D, hout);

        bn_stats<<<NN / BN_ROWS, 256, 0, stream>>>(hout, sums, sqs);
        bn_apply<<<egrid, 256, 0, stream>>>(hout, sums, sqs,
                                            gamma + (size_t)l * D, beta + (size_t)l * D,
                                            h, (l < 4) ? 1 : 0);
    }
}
